// SelfAttention_15822659518595
// MI455X (gfx1250) — matrix operand order repaired and checked by reading, not compiled, not run on hardware
//
#include <hip/hip_runtime.h>
#include <hip/hip_bf16.h>

// ---------------------------------------------------------------------------
// Self-attention (B=8, S=2048, D=1024) for MI455X / gfx1250, wave32 + WMMA.
//
//   0) x,W -> bf16 (elementwise pre-pass; removes VALU cvt from GEMM mainloop)
//   1) qkv = x @ W^T + b       bf16 WMMA GEMM, epilogue emits qkv(bf16) + V^T
//   2) S   = q @ k^T / D       bf16 WMMA GEMM (batched), fp32 scores
//   3) column stats over i     (softmax is over axis=1 = query axis!)
//   4) P   = exp(S - m_j)/Z_j  -> bf16
//   5) out = P @ vt^T          bf16 WMMA GEMM (batched), fp32 out
//
// GEMM mainloop: 128x256 block tile, 8 waves x (64x64), K-step 32 ->
// 16 WMMAs per 16 ds_load_b128 per wave per K-step. Tile staging uses the
// Tensor Data Mover (tensor_load_to_lds, D# built per ISA ch.8 with HW LDS
// row padding), double buffered, synchronized via s_wait_tensorcnt; falls
// back to vector loads + ds_store if the builtin is absent.
//
// scores (128MiB) fit in the 192MiB L2 -> passes 2..5 are L2-resident.
// ---------------------------------------------------------------------------

#define S_LEN   2048
#define D_MODEL 1024
#define NBATCH  8

typedef __attribute__((ext_vector_type(16))) __bf16 v16bf;
typedef __attribute__((ext_vector_type(8)))  float  v8f;
typedef unsigned int u32x4 __attribute__((ext_vector_type(4)));
typedef int          i32x4 __attribute__((ext_vector_type(4)));
typedef int          i32x8 __attribute__((ext_vector_type(8)));

#if defined(__has_builtin)
#  if __has_builtin(__builtin_amdgcn_tensor_load_to_lds)
#    define ATHENA_HAS_TDM 1
#  endif
#  if __has_builtin(__builtin_amdgcn_s_wait_tensorcnt)
#    define WAIT_TENSORCNT(n) __builtin_amdgcn_s_wait_tensorcnt((short)(n))
#  endif
#endif
#ifndef WAIT_TENSORCNT
#  define WAIT_TENSORCNT(n) asm volatile("s_wait_tensorcnt %0" ::"i"(n) : "memory")
#endif
#if defined(ATHENA_HAS_TDM) && __has_include(<hip/amd_detail/amd_gfx1250_TDM.h>)
#  define ATHENA_TDM_6ARG 1
#endif

__device__ __forceinline__ unsigned short f2b(float f) {
  // round-to-nearest-even fp32 -> bf16
  unsigned u = __builtin_bit_cast(unsigned, f);
  u += 0x7FFFu + ((u >> 16) & 1u);
  return (unsigned short)(u >> 16);
}

union FragU { uint4 u[2]; v16bf v; };

#ifdef ATHENA_HAS_TDM
// TDM load of one rows x 32(cols) bf16 tile into LDS at lds_off.
// D# per ISA ch.8: 2D tile, data_size=1 (2B); pad_enable with
// pad_interval=3 (every 16 DWORDs = one 64B row) and pad_amount=3 (4 DWORDs
// = 16B) -> LDS row stride 80B = 40 elements (matches fragment addressing).
__device__ __forceinline__ void tdm_tile_load(const unsigned short* gtile,
                                              unsigned lds_off, long ld_elems,
                                              unsigned rows) {
  unsigned long long ga = (unsigned long long)(size_t)gtile;
  u32x4 g0;
  g0[0] = 1u;                                    // count=1, user mode
  g0[1] = lds_off;                               // lds_addr (bytes)
  g0[2] = (unsigned)(ga & 0xFFFFFFFFu);          // global_addr[31:0]
  g0[3] = (unsigned)((ga >> 32) & 0x1FFFFFFu)    // global_addr[56:32]
          | (2u << 30);                          // type = 2 ("image")
  const unsigned td0 = 1u << 20, td1 = 1u << 20; // large dims: no OOB clip
  const unsigned tile0 = 32;
  unsigned long long s0 = (unsigned long long)ld_elems;
  i32x8 g1;
  g1[0] = (int)((1u << 16)          // data_size = 1 (2 bytes)
              | (1u << 20)          // pad_enable
              | (3u << 22)          // pad_interval: 16 DWORDs
              | (3u << 25));        // pad_amount : 4 DWORDs
  g1[1] = (int)((td0 & 0xFFFFu) << 16);                  // tensor_dim0[15:0]
  g1[2] = (int)((td0 >> 16) | ((td1 & 0xFFFFu) << 16));  // td0 hi | td1 lo
  g1[3] = (int)((td1 >> 16) | (tile0 << 16));            // td1 hi | tile_dim0
  g1[4] = (int)rows;                                     // tile_dim1 (dim2=0)
  g1[5] = (int)(s0 & 0xFFFFFFFFu);                       // dim0_stride[31:0]
  g1[6] = (int)((s0 >> 32) & 0xFFFFu);                   // dim0_stride[47:32]
  g1[7] = 0;
  i32x4 z4 = {0, 0, 0, 0};
#ifdef ATHENA_TDM_6ARG
  i32x8 z8 = {0, 0, 0, 0, 0, 0, 0, 0};
  __builtin_amdgcn_tensor_load_to_lds(g0, g1, z4, z4, z8, 0);
#else
  __builtin_amdgcn_tensor_load_to_lds(g0, g1, z4, z4, 0);
#endif
}
#endif  // ATHENA_HAS_TDM

// C[M,N] = A[M,K] * B[N,K]^T, both operands bf16 row-major over K.
// mode 0: C += bias[n] -> bf16 Cbf (ld=3072); cols>=2048 also -> vt[b][d][j].
// mode 1: Cf = C * scale   mode 2: Cf = C
__global__ __launch_bounds__(256) void gemm_bf16_wmma(
    const unsigned short* __restrict__ A, long lda, long strideA,
    const unsigned short* __restrict__ B, long ldb, long strideB,
    int M, int N, int K,
    int mode, const float* __restrict__ bias, float scale,
    float* __restrict__ Cf, unsigned short* __restrict__ Cbf,
    unsigned short* __restrict__ vt,
    long ldc, long strideC)
{
  // double-buffered tiles, rows padded to 40 bf16 (80B)
  __shared__ __align__(16) unsigned short a_tile[2][128 * 40];  // 2 x 10KB
  __shared__ __align__(16) unsigned short b_tile[2][256 * 40];  // 2 x 20KB

  const int tid  = threadIdx.x;
  const int lane = tid & 31;
  const int wave = tid >> 5;
  const int wm   = wave & 1;   // 2 waves along M -> 64 rows each
  const int wn   = wave >> 1;  // 4 waves along N -> 64 cols each
  const int ln   = lane & 15;
  const int h    = lane >> 4;

  const long m0 = (long)blockIdx.y * 128;
  const long n0 = (long)blockIdx.x * 256;
  const long z  = blockIdx.z;

  const unsigned short* Ab = A + z * strideA + m0 * lda;
  const unsigned short* Bb = B + z * strideB + n0 * ldb;
  float* Cfb = Cf ? Cf + z * strideC : nullptr;

  v8f acc[4][4] = {};
  const int KB = K >> 5;

#ifdef ATHENA_HAS_TDM
  const unsigned a_off[2] = {(unsigned)(size_t)&a_tile[0][0],
                             (unsigned)(size_t)&a_tile[1][0]};
  const unsigned b_off[2] = {(unsigned)(size_t)&b_tile[0][0],
                             (unsigned)(size_t)&b_tile[1][0]};
  if (wave == 0) {                       // wave 0 drives the DMA engine
    tdm_tile_load(Ab, a_off[0], lda, 128);
    tdm_tile_load(Bb, b_off[0], ldb, 256);
  }
#endif

  for (int kb = 0; kb < KB; ++kb) {
    const int buf = kb & 1;
#ifdef ATHENA_HAS_TDM
    if (wave == 0) {
      if (kb + 1 < KB) {                 // prefetch next tile pair, then wait
        tdm_tile_load(Ab + (long)(kb + 1) * 32, a_off[buf ^ 1], lda, 128);
        tdm_tile_load(Bb + (long)(kb + 1) * 32, b_off[buf ^ 1], ldb, 256);
        WAIT_TENSORCNT(2);               // current pair has landed in LDS
      } else {
        WAIT_TENSORCNT(0);
      }
    }
#else
    // fallback: vector-load + ds_store staging (all 256 threads)
    {
      const int k0 = kb * 32;
#pragma unroll
      for (int it = 0; it < 2; ++it) {   // A: 512 chunks of 8 bf16
        int chunk = tid + it * 256;
        int r  = chunk >> 2;
        int c8 = (chunk & 3) << 3;
        *(uint4*)&a_tile[buf][r * 40 + c8] =
            *(const uint4*)(Ab + (long)r * lda + k0 + c8);
      }
#pragma unroll
      for (int it = 0; it < 4; ++it) {   // B: 1024 chunks of 8 bf16
        int chunk = tid + it * 256;
        int r  = chunk >> 2;
        int c8 = (chunk & 3) << 3;
        *(uint4*)&b_tile[buf][r * 40 + c8] =
            *(const uint4*)(Bb + (long)r * ldb + k0 + c8);
      }
    }
#endif
    __syncthreads();                     // tiles visible to all waves

    FragU bfrag[4];
#pragma unroll
    for (int nt = 0; nt < 4; ++nt) {
      int c = wn * 64 + nt * 16 + ln;
      const unsigned short* p = &b_tile[buf][c * 40 + h * 16];
      bfrag[nt].u[0] = *(const uint4*)(p);
      bfrag[nt].u[1] = *(const uint4*)(p + 8);
    }
#pragma unroll
    for (int mt = 0; mt < 4; ++mt) {
      int r = wm * 64 + mt * 16 + ln;
      const unsigned short* p = &a_tile[buf][r * 40 + h * 16];
      FragU af;
      af.u[0] = *(const uint4*)(p);
      af.u[1] = *(const uint4*)(p + 8);
#pragma unroll
      for (int nt = 0; nt < 4; ++nt) {
        acc[mt][nt] = __builtin_amdgcn_wmma_f32_16x16x32_bf16(
            false, af.v, false, bfrag[nt].v, (short)0, acc[mt][nt],
            false, false);
      }
    }
    __syncthreads();                     // done reading buf before overwrite
  }

  // epilogue: C/D f32 layout: reg v -> row +v+8h, lane -> col
#pragma unroll
  for (int mt = 0; mt < 4; ++mt) {
#pragma unroll
    for (int nt = 0; nt < 4; ++nt) {
      long row0 = m0 + wm * 64 + mt * 16 + h * 8;
      long col  = n0 + wn * 64 + nt * 16 + ln;
#pragma unroll
      for (int r = 0; r < 8; ++r) {
        long row = row0 + r;
        float val = acc[mt][nt][r];
        if (mode == 0) {
          val += bias[col];
          unsigned short bv = f2b(val);
          Cbf[row * ldc + col] = bv;
          if (col >= 2 * D_MODEL) {
            long bb = row >> 11;           // / S_LEN
            long j  = row & (S_LEN - 1);
            vt[(bb * D_MODEL + (col - 2 * D_MODEL)) * S_LEN + j] = bv;
          }
        } else {
          Cfb[row * ldc + col] = val * scale;
        }
      }
    }
  }
}

// fp32 -> bf16 elementwise (8 elements / thread)
__global__ __launch_bounds__(256) void f32_to_bf16_kernel(
    const float* __restrict__ src, unsigned short* __restrict__ dst)
{
  long g = ((long)blockIdx.x * 256 + threadIdx.x) * 8;
  float4 f0 = *(const float4*)(src + g);
  float4 f1 = *(const float4*)(src + g + 4);
  uint4 d;
  d.x = (unsigned)f2b(f0.x) | ((unsigned)f2b(f0.y) << 16);
  d.y = (unsigned)f2b(f0.z) | ((unsigned)f2b(f0.w) << 16);
  d.z = (unsigned)f2b(f1.x) | ((unsigned)f2b(f1.y) << 16);
  d.w = (unsigned)f2b(f1.z) | ((unsigned)f2b(f1.w) << 16);
  *(uint4*)(dst + g) = d;
}

// Per-column (axis=1 = query axis) online max / sum-exp over i.
__global__ __launch_bounds__(256) void colstats_kernel(
    const float* __restrict__ s, float* __restrict__ mcol,
    float* __restrict__ zinv)
{
  int g = blockIdx.x * 256 + threadIdx.x;   // [0, NBATCH*S_LEN)
  int b = g >> 11;
  int j = g & (S_LEN - 1);
  const float* col = s + (long)b * S_LEN * S_LEN + j;
  float m = -3.402823466e38f, l = 0.0f;
  for (int i = 0; i < S_LEN; ++i) {
    float v  = col[(long)i * S_LEN];
    float nm = fmaxf(m, v);
    l = l * __expf(m - nm) + __expf(v - nm);
    m = nm;
  }
  mcol[g] = m;
  zinv[g] = 1.0f / l;
}

// P[b,i,j] = exp(S - m_j) * zinv_j -> bf16 (A operand of the output GEMM)
__global__ __launch_bounds__(256) void softmax_norm_kernel(
    const float* __restrict__ s, const float* __restrict__ mcol,
    const float* __restrict__ zinv, unsigned short* __restrict__ p)
{
  long g = ((long)blockIdx.x * 256 + threadIdx.x) * 4;
  long b   = g >> 22;                       // / (S*S)
  long rem = g & ((1L << 22) - 1);
  long j   = rem & (S_LEN - 1);
  long mz  = (b << 11) + j;
  float4 v = *(const float4*)(s + g);
  float m0_ = mcol[mz],     m1 = mcol[mz + 1];
  float m2  = mcol[mz + 2], m3 = mcol[mz + 3];
  float z0  = zinv[mz],     z1 = zinv[mz + 1];
  float z2  = zinv[mz + 2], z3 = zinv[mz + 3];
  unsigned short h0 = f2b(__expf(v.x - m0_) * z0);
  unsigned short h1 = f2b(__expf(v.y - m1)  * z1);
  unsigned short h2 = f2b(__expf(v.z - m2)  * z2);
  unsigned short h3 = f2b(__expf(v.w - m3)  * z3);
  uint2 d;
  d.x = (unsigned)h0 | ((unsigned)h1 << 16);
  d.y = (unsigned)h2 | ((unsigned)h3 << 16);
  *(uint2*)(p + g) = d;
}

extern "C" void kernel_launch(void* const* d_in, const int* in_sizes, int n_in,
                              void* d_out, int out_size, void* d_ws, size_t ws_size,
                              hipStream_t stream) {
  const float* x    = (const float*)d_in[0];   // [8, 2048, 1024]
  const float* W    = (const float*)d_in[1];   // [3072, 1024]
  const float* bias = (const float*)d_in[2];   // [3072]
  float* out = (float*)d_out;                  // [8, 2048, 1024]

  const long M  = (long)NBATCH * S_LEN;        // 16384
  const long N3 = 3 * D_MODEL;                 // 3072
  const long SS = (long)S_LEN * S_LEN;

  char* w = (char*)d_ws;
  unsigned short* qkv = (unsigned short*)w;  w += M * N3 * 2;                            // 96MiB
  unsigned short* vt  = (unsigned short*)w;  w += (long)NBATCH * D_MODEL * S_LEN * 2;    // 32MiB
  float* scores       = (float*)w;           w += (long)NBATCH * SS * 4;                 // 128MiB
  unsigned short* p   = (unsigned short*)w;  w += (long)NBATCH * SS * 2;                 // 64MiB
  unsigned short* xb  = (unsigned short*)w;  w += M * D_MODEL * 2;                       // 32MiB
  unsigned short* Wb  = (unsigned short*)w;  w += N3 * D_MODEL * 2;                      // 6MiB
  float* mcol         = (float*)w;           w += M * 4;
  float* zinv         = (float*)w;

  dim3 blk(256);

  // 0) convert x, W to bf16
  f32_to_bf16_kernel<<<dim3((unsigned)(M * D_MODEL / 2048)), blk, 0, stream>>>(x, xb);
  f32_to_bf16_kernel<<<dim3((unsigned)(N3 * D_MODEL / 2048)), blk, 0, stream>>>(W, Wb);

  // 1) QKV projection: [16384,3072] = x[16384,1024] * W[3072,1024]^T + b
  gemm_bf16_wmma<<<dim3(N3 / 256, M / 128, 1), blk, 0, stream>>>(
      xb, D_MODEL, 0, Wb, D_MODEL, 0,
      (int)M, (int)N3, D_MODEL,
      /*mode=*/0, bias, 1.0f, nullptr, qkv, vt, N3, 0);

  // 2) scores[b] = q[b] * k[b]^T / D  (q = qkv cols 0..1023, k = 1024..2047)
  gemm_bf16_wmma<<<dim3(S_LEN / 256, S_LEN / 128, NBATCH), blk, 0, stream>>>(
      qkv, N3, (long)S_LEN * N3,
      qkv + D_MODEL, N3, (long)S_LEN * N3,
      S_LEN, S_LEN, D_MODEL,
      /*mode=*/1, nullptr, 1.0f / (float)D_MODEL, scores, nullptr, nullptr,
      S_LEN, SS);

  // 3) per-column softmax stats (axis=1)
  colstats_kernel<<<dim3((NBATCH * S_LEN) / 256), blk, 0, stream>>>(scores, mcol, zinv);

  // 4) normalize -> bf16 P
  softmax_norm_kernel<<<dim3((unsigned)((NBATCH * SS) / 1024)), blk, 0, stream>>>(
      scores, mcol, zinv, p);

  // 5) out[b] = P[b] * vt[b]^T
  gemm_bf16_wmma<<<dim3(D_MODEL / 256, S_LEN / 128, NBATCH), blk, 0, stream>>>(
      p, S_LEN, SS,
      vt, S_LEN, (long)D_MODEL * S_LEN,
      S_LEN, D_MODEL, S_LEN,
      /*mode=*/2, nullptr, 1.0f, out, nullptr, nullptr,
      D_MODEL, (long)S_LEN * D_MODEL);
}
